// RegionSABIKAN_84756884619573
// MI455X (gfx1250) — compile-verified, matrix operand order checked by reading, not run on hardware
//
#include <hip/hip_runtime.h>
#include <math.h>

// ---------------------------------------------------------------------------
// Shapes (fixed for this problem)
// ---------------------------------------------------------------------------
#define B_    16
#define C_    256          // CIN == COUT
#define P_    4096         // H*W
#define GNC   131072.0f    // elements per group-norm group: 32 * 4096
#define SLAB  ((size_t)C_ * (size_t)P_)   // per-batch plane elements
#define PADW  132          // padded LDS row stride (floats); 132 % 64 = 4 -> no bank conflicts

typedef __attribute__((ext_vector_type(16))) __bf16 bf16x16;
typedef __attribute__((ext_vector_type(8)))  float  f32x8;
typedef __attribute__((ext_vector_type(4)))  int    i32x4;

#ifndef __has_builtin
#define __has_builtin(x) 0
#endif
#if __has_builtin(__builtin_amdgcn_global_load_async_to_lds_b128) && \
    __has_builtin(__builtin_amdgcn_s_wait_asynccnt)
#define USE_ASYNC_LDS 1
#else
#define USE_ASYNC_LDS 0
#endif

#if USE_ASYNC_LDS
#define ASYNC_WAIT(n) __builtin_amdgcn_s_wait_asynccnt(n)
#else
#define ASYNC_WAIT(n)
#endif

__device__ __forceinline__ float sigmoidf_(float x) { return 1.f / (1.f + __expf(-x)); }
__device__ __forceinline__ float geluf_(float x) {
  // tanh-approx gelu (jax.nn.gelu default)
  float x3 = x * x * x;
  return 0.5f * x * (1.f + tanhf(0.79788456080286535588f * (x + 0.044715f * x3)));
}

// One 16-byte global->LDS copy (async engine when available).
__device__ __forceinline__ void copy16_g2l(const float* g, float* l) {
#if USE_ASYNC_LDS
  __builtin_amdgcn_global_load_async_to_lds_b128(
      (__attribute__((address_space(1))) i32x4*)(uintptr_t)g,
      (__attribute__((address_space(3))) i32x4*)(uint32_t)(uintptr_t)l,
      0, 0);
#else
  *(float4*)l = *(const float4*)g;
#endif
}

// ---------------------------------------------------------------------------
// Kernel 0: zero the small stats region (512 floats: pre + post stats)
// ---------------------------------------------------------------------------
__global__ void zero_small(float* __restrict__ p) { p[threadIdx.x] = 0.f; }

// ---------------------------------------------------------------------------
// Kernel 1/5: Z[b][o][p] = sum_c W[o][c] * X[b][c][p] + bias[o]
// v_wmma_f32_16x16x32_bf16; X K-panel staged in LDS via async-to-LDS,
// double buffered; f32->bf16 on the fly, f32 accumulate.
// Also accumulates per-(b, gn-group) sum/sumsq atomics for group-norm.
// grid = (P/128, COUT/64, B), block = 256 (8 waves: 4 M-subtiles x 2 N-groups)
// ---------------------------------------------------------------------------
__global__ __launch_bounds__(256)
void gemm_bias_stats(const float* __restrict__ X, const float* __restrict__ Wm,
                     const float* __restrict__ bias, float* __restrict__ Z,
                     float* __restrict__ stats /* [B][8][2] */)
{
  __shared__ float Xs[2][32 * PADW];   // double-buffered 32x128 f32 K-panel (padded)

  const int tid   = threadIdx.x;
  const int lane  = tid & 31;
  const int wave  = tid >> 5;
  const int wm    = wave & 3;        // M sub-tile within block
  const int wn    = wave >> 2;       // N half within block
  const int b     = blockIdx.z;
  const int otile = blockIdx.y * 64 + wm * 16;
  const int pblk  = blockIdx.x * 128;
  const int ln    = lane & 15;       // 0..15
  const int h     = lane >> 4;       // 0..1 (K-half selector per ISA layout)

  const float* Xb = X + (size_t)b * SLAB + pblk;

  // staging coords: thread copies rows {srow, srow+8, srow+16, srow+24}, 4 floats each
  const int srow = tid >> 5;         // 0..7
  const int scol = (tid & 31) * 4;   // 0..124

  f32x8 acc0 = {}, acc1 = {}, acc2 = {}, acc3 = {};

  // prefetch K-panel 0
#pragma unroll
  for (int rr = 0; rr < 4; ++rr) {
    const int row = srow + rr * 8;
    copy16_g2l(Xb + (size_t)row * P_ + scol, &Xs[0][row * PADW + scol]);
  }

  for (int it = 0; it < 8; ++it) {
    const int cur = it & 1;
    if (it < 7) {
      // prefetch next K-panel into the other buffer
#pragma unroll
      for (int rr = 0; rr < 4; ++rr) {
        const int row = srow + rr * 8;
        copy16_g2l(Xb + (size_t)((it + 1) * 32 + row) * P_ + scol,
                   &Xs[cur ^ 1][row * PADW + scol]);
      }
      ASYNC_WAIT(4);   // retire the 4 copies of panel `it` (async loads complete in order)
    } else {
      ASYNC_WAIT(0);
    }
    __syncthreads();   // panel `cur` visible to all waves

    const int kt = it * 32;

    // ---- A fragment: 16x32 bf16 of W, row = otile+ln (global; L2-resident) ----
    // lanes 0-15 hold K = {0..7, 16..23}; lanes 16-31 hold K = {8..15, 24..31}
    bf16x16 afr;
    {
      const float* wr = Wm + (size_t)(otile + ln) * C_ + kt + h * 8;
#pragma unroll
      for (int r = 0; r < 8; ++r) {
        const int kb = ((r < 4) ? 0 : 16) + (r & 3) * 2;
        float2 wv = *(const float2*)(wr + kb);
        afr[2 * r]     = (__bf16)wv.x;
        afr[2 * r + 1] = (__bf16)wv.y;
      }
    }
    // ---- B fragments from LDS: four 32x16 tiles, col = wn*64 + t*16 + ln ----
    const float* xs = &Xs[cur][0];
    bf16x16 bfr0, bfr1, bfr2, bfr3;
#pragma unroll
    for (int r = 0; r < 8; ++r) {
      const int kb = ((r < 4) ? 0 : 16) + (r & 3) * 2 + h * 8;
      const float* p0 = xs + kb * PADW + wn * 64 + ln;  // K row kb
      const float* p1 = p0 + PADW;                      // K row kb+1
      bfr0[2 * r] = (__bf16)p0[0];  bfr0[2 * r + 1] = (__bf16)p1[0];
      bfr1[2 * r] = (__bf16)p0[16]; bfr1[2 * r + 1] = (__bf16)p1[16];
      bfr2[2 * r] = (__bf16)p0[32]; bfr2[2 * r + 1] = (__bf16)p1[32];
      bfr3[2 * r] = (__bf16)p0[48]; bfr3[2 * r + 1] = (__bf16)p1[48];
    }
    acc0 = __builtin_amdgcn_wmma_f32_16x16x32_bf16(false, afr, false, bfr0, (short)0, acc0, false, false);
    acc1 = __builtin_amdgcn_wmma_f32_16x16x32_bf16(false, afr, false, bfr1, (short)0, acc1, false, false);
    acc2 = __builtin_amdgcn_wmma_f32_16x16x32_bf16(false, afr, false, bfr2, (short)0, acc2, false, false);
    acc3 = __builtin_amdgcn_wmma_f32_16x16x32_bf16(false, afr, false, bfr3, (short)0, acc3, false, false);

    __syncthreads();   // all waves done reading panel `cur` before it is overwritten
  }

  // ---- Epilogue: bias, store, group-norm partial stats ------------------
  // D layout: element r of acc in lane L -> m = r + 8*(L>>4), n = L&15
  float bsum = 0.f, bsq = 0.f;
#pragma unroll
  for (int t = 0; t < 4; ++t) {
    const f32x8 a = (t == 0) ? acc0 : (t == 1) ? acc1 : (t == 2) ? acc2 : acc3;
#pragma unroll
    for (int r = 0; r < 8; ++r) {
      const int o = otile + r + 8 * h;
      const int n = pblk + wn * 64 + t * 16 + ln;
      float v = a[r] + bias[o];
      Z[(size_t)b * SLAB + (size_t)o * P_ + n] = v;
      bsum += v;
      bsq  += v * v;
    }
  }
#pragma unroll
  for (int off = 16; off; off >>= 1) {
    bsum += __shfl_xor(bsum, off, 32);
    bsq  += __shfl_xor(bsq,  off, 32);
  }
  if (lane == 0) {
    const int g = otile >> 5;  // 16-row tile lies in one 32-channel group
    atomicAdd(&stats[(b * 8 + g) * 2 + 0], bsum);
    atomicAdd(&stats[(b * 8 + g) * 2 + 1], bsq);
  }
}

// ---------------------------------------------------------------------------
// Kernel 2/6: group-norm + SiLU (elementwise given stats); optional pooling.
// One block per (b,o) row of 4096 pixels.
// ---------------------------------------------------------------------------
__global__ __launch_bounds__(256)
void gn_silu_pool(const float* __restrict__ Zin, float* __restrict__ Zout,
                  const float* __restrict__ stats, const float* __restrict__ gamma,
                  const float* __restrict__ beta, float* __restrict__ pooled,
                  int doPool)
{
  const int bo = blockIdx.x;
  const int b = bo >> 8, o = bo & 255;
  const int g = o >> 5;
  const float inv  = 1.0f / GNC;
  const float mean = stats[(b * 8 + g) * 2 + 0] * inv;
  const float var  = stats[(b * 8 + g) * 2 + 1] * inv - mean * mean;
  const float rstd = rsqrtf(var + 1e-5f);
  const float ga = gamma[o] * rstd;
  const float be = beta[o] - mean * ga;

  const float4* src = (const float4*)(Zin + (size_t)bo * P_);
  float4* dst = (float4*)(Zout + (size_t)bo * P_);

  float psum = 0.f;
  for (int i = threadIdx.x; i < P_ / 4; i += 256) {
    float4 v = src[i];
    float y0 = v.x * ga + be, y1 = v.y * ga + be;
    float y2 = v.z * ga + be, y3 = v.w * ga + be;
    float s0 = y0 * sigmoidf_(y0), s1 = y1 * sigmoidf_(y1);
    float s2 = y2 * sigmoidf_(y2), s3 = y3 * sigmoidf_(y3);
    dst[i] = make_float4(s0, s1, s2, s3);
    psum += s0 + s1 + s2 + s3;
  }
  if (doPool) {
    const int lane = threadIdx.x & 31, wave = threadIdx.x >> 5;
#pragma unroll
    for (int off = 16; off; off >>= 1) psum += __shfl_xor(psum, off, 32);
    __shared__ float red[8];
    if (lane == 0) red[wave] = psum;
    __syncthreads();
    if (threadIdx.x == 0) {
      float t = 0.f;
#pragma unroll
      for (int i = 0; i < 8; ++i) t += red[i];
      pooled[bo] = t * (1.0f / (float)P_);
    }
  }
}

// ---------------------------------------------------------------------------
// Kernel 3: squeeze MLP + heads (single workgroup; tiny problem).
// params layout (floats): alpha[384] | mu[384] | sigma[384] | mix[384] | gate[64]
// ---------------------------------------------------------------------------
__global__ __launch_bounds__(256)
void mlp_heads(const float* __restrict__ pooled,
               const float* __restrict__ w1, const float* __restrict__ b1,
               const float* __restrict__ w2, const float* __restrict__ b2,
               const float* __restrict__ aw, const float* __restrict__ ab,
               const float* __restrict__ mw, const float* __restrict__ mb,
               const float* __restrict__ sw, const float* __restrict__ sb,
               const float* __restrict__ xw, const float* __restrict__ xb,
               const float* __restrict__ gw, const float* __restrict__ gb,
               float* __restrict__ params)
{
  __shared__ float pl[16 * 256];
  __shared__ float t1[16 * 64];
  __shared__ float sv[16 * 64];
  __shared__ float araw[16 * 24];
  const int tid = threadIdx.x;

  for (int i = tid; i < 16 * 256; i += 256) pl[i] = pooled[i];
  __syncthreads();

  for (int i = tid; i < 16 * 64; i += 256) {
    const int bb = i >> 6, j = i & 63;
    float acc = b1[j];
    const float* pr = &pl[bb * 256];
    for (int c = 0; c < 256; ++c) acc += pr[c] * w1[c * 64 + j];
    t1[i] = geluf_(acc);
  }
  __syncthreads();

  for (int i = tid; i < 16 * 64; i += 256) {
    const int bb = i >> 6, j = i & 63;
    float acc = b2[j];
    const float* tr = &t1[bb * 64];
    for (int k = 0; k < 64; ++k) acc += tr[k] * w2[k * 64 + j];
    sv[i] = geluf_(acc);
  }
  __syncthreads();

  // 4 heads of width G*NB = 24
  for (int i = tid; i < 16 * 24; i += 256) {
    const int bb = i / 24, m = i % 24;
    const float* sr = &sv[bb * 64];
    float a = ab[m], u = mb[m], s = sb[m], x = xb[m];
    for (int j = 0; j < 64; ++j) {
      const float sj = sr[j];
      a += sj * aw[j * 24 + m];
      u += sj * mw[j * 24 + m];
      s += sj * sw[j * 24 + m];
      x += sj * xw[j * 24 + m];
    }
    araw[i] = a;
    params[384 + i]  = u;                                      // mu
    float sp = (s > 20.f) ? s : log1pf(__expf(s));             // softplus
    params[768 + i]  = sp + 0.001f;                            // sigma
    params[1152 + i] = sigmoidf_(x);                           // mix
  }
  // gate head (width G = 4)
  for (int i = tid; i < 64; i += 256) {
    const int bb = i >> 2, gg = i & 3;
    const float* sr = &sv[bb * 64];
    float a = gb[gg];
    for (int j = 0; j < 64; ++j) a += sr[j] * gw[j * 4 + gg];
    params[1536 + i] = sigmoidf_(a);
  }
  __syncthreads();

  // softmax over NB=6 per (b,g)
  for (int i = tid; i < 64; i += 256) {
    const float* ar = &araw[i * 6];
    float mx = ar[0];
#pragma unroll
    for (int k = 1; k < 6; ++k) mx = fmaxf(mx, ar[k]);
    float e[6], ssum = 0.f;
#pragma unroll
    for (int k = 0; k < 6; ++k) { e[k] = __expf(ar[k] - mx); ssum += e[k]; }
    const float r = 1.f / ssum;
#pragma unroll
    for (int k = 0; k < 6; ++k) params[i * 6 + k] = e[k] * r;
  }
}

// ---------------------------------------------------------------------------
// Kernel 4: Gaussian-mixture response.  grid = (64 chunks, 64 (b,g) slabs).
// y = z + gate * sum_nb alpha*(mix*eu + (1-mix)*u*eu), u=(z-mu)/sigma, eu=e^-u^2
// ---------------------------------------------------------------------------
__global__ __launch_bounds__(256)
void response_kernel(const float* __restrict__ Z, float* __restrict__ Y,
                     const float* __restrict__ params)
{
  const int bg = blockIdx.y;           // b*4 + g
  __shared__ float al[6], muv[6], sg[6], mx[6];
  __shared__ float gatev;
  if (threadIdx.x < 6) {
    al[threadIdx.x]  = params[bg * 6 + threadIdx.x];
    muv[threadIdx.x] = params[384 + bg * 6 + threadIdx.x];
    sg[threadIdx.x]  = params[768 + bg * 6 + threadIdx.x];
    mx[threadIdx.x]  = params[1152 + bg * 6 + threadIdx.x];
  }
  if (threadIdx.x == 6) gatev = params[1536 + bg];
  __syncthreads();

  const float* src = Z + (size_t)bg * 64 * P_;  // (b*256 + g*64)*P_
  float* dst       = Y + (size_t)bg * 64 * P_;
  const size_t base = (size_t)blockIdx.x * 4096;

  for (int i = threadIdx.x; i < 1024; i += 256) {
    float4 z4 = ((const float4*)(src + base))[i];
    float out[4];
    const float zin[4] = {z4.x, z4.y, z4.z, z4.w};
#pragma unroll
    for (int c = 0; c < 4; ++c) {
      const float z = zin[c];
      float resp = 0.f;
#pragma unroll
      for (int k = 0; k < 6; ++k) {
        const float u  = (z - muv[k]) / sg[k];
        const float eu = __expf(-u * u);
        resp += al[k] * (mx[k] * eu + (1.f - mx[k]) * (u * eu));
      }
      out[c] = z + gatev * resp;
    }
    ((float4*)(dst + base))[i] = make_float4(out[0], out[1], out[2], out[3]);
  }
}

// ---------------------------------------------------------------------------
// Launch
// ---------------------------------------------------------------------------
extern "C" void kernel_launch(void* const* d_in, const int* in_sizes, int n_in,
                              void* d_out, int out_size, void* d_ws, size_t ws_size,
                              hipStream_t stream)
{
  (void)in_sizes; (void)n_in; (void)out_size; (void)ws_size;
  const float* x         = (const float*)d_in[0];
  const float* pre_w     = (const float*)d_in[1];
  const float* pre_b     = (const float*)d_in[2];
  const float* pre_gn_g  = (const float*)d_in[3];
  const float* pre_gn_b  = (const float*)d_in[4];
  const float* sp_w1     = (const float*)d_in[5];
  const float* sp_b1     = (const float*)d_in[6];
  const float* sp_w2     = (const float*)d_in[7];
  const float* sp_b2     = (const float*)d_in[8];
  const float* alpha_w   = (const float*)d_in[9];
  const float* alpha_b   = (const float*)d_in[10];
  const float* mu_w      = (const float*)d_in[11];
  const float* mu_b      = (const float*)d_in[12];
  const float* sigma_w   = (const float*)d_in[13];
  const float* sigma_b   = (const float*)d_in[14];
  const float* mix_w     = (const float*)d_in[15];
  const float* mix_b     = (const float*)d_in[16];
  const float* gate_w    = (const float*)d_in[17];
  const float* gate_b    = (const float*)d_in[18];
  const float* post_w    = (const float*)d_in[19];
  const float* post_b    = (const float*)d_in[20];
  const float* post_gn_g = (const float*)d_in[21];
  const float* post_gn_b = (const float*)d_in[22];
  float* out = (float*)d_out;

  float* bufA       = (float*)d_ws;                       // 64 MB: z_pre -> z -> y2_pre
  float* bufB       = bufA + (size_t)B_ * SLAB;           // 64 MB: y
  float* stats_pre  = bufB + (size_t)B_ * SLAB;           // 256 floats
  float* stats_post = stats_pre + 256;                    // 256 floats
  float* pooled     = stats_post + 256;                   // 4096 floats
  float* params     = pooled + 4096;                      // 1600 floats

  zero_small<<<1, 512, 0, stream>>>(stats_pre);           // clears pre+post stats

  gemm_bias_stats<<<dim3(32, 4, 16), 256, 0, stream>>>(x, pre_w, pre_b, bufA, stats_pre);
  gn_silu_pool<<<4096, 256, 0, stream>>>(bufA, bufA, stats_pre, pre_gn_g, pre_gn_b, pooled, 1);
  mlp_heads<<<1, 256, 0, stream>>>(pooled, sp_w1, sp_b1, sp_w2, sp_b2,
                                   alpha_w, alpha_b, mu_w, mu_b, sigma_w, sigma_b,
                                   mix_w, mix_b, gate_w, gate_b, params);
  response_kernel<<<dim3(64, 64), 256, 0, stream>>>(bufA, bufB, params);
  gemm_bias_stats<<<dim3(32, 4, 16), 256, 0, stream>>>(bufB, post_w, post_b, bufA, stats_post);
  gn_silu_pool<<<4096, 256, 0, stream>>>(bufA, out, stats_post, post_gn_g, post_gn_b, pooled, 0);
}
